// CrossAttn_48558900248806
// MI455X (gfx1250) — matrix operand order repaired and checked by reading, not compiled, not run on hardware
//
#include <hip/hip_runtime.h>
#include <hip/hip_bf16.h>
#include <stdint.h>

typedef __attribute__((ext_vector_type(16))) __bf16 v16bf;
typedef __attribute__((ext_vector_type(8)))  float  v8f;

namespace {
constexpr int Bc = 4, Cc = 48, Hh = 128, Ww = 128;
constexpr int DIM  = 384;        // N*C
constexpr int QKVC = 3 * DIM;    // 1152
constexpr int HW   = Hh * Ww;    // 16384
constexpr int MT = 128, NTB = 64, KT = 32;
constexpr int NK = DIM / KT;     // 12
constexpr int LDA = 72;          // halves per A row in LDS (padded, 144B stride)
constexpr int LDB = 40;          // halves per B row in LDS (padded, 80B stride)

__device__ __forceinline__ unsigned short f2bf(float f) {
  return __builtin_bit_cast(unsigned short, (__bf16)f);   // hw cvt, RTNE
}
__device__ __forceinline__ unsigned pk2f(float a, float b) {
  unsigned short ua = __builtin_bit_cast(unsigned short, (__bf16)a);
  unsigned short ub = __builtin_bit_cast(unsigned short, (__bf16)b);
  return (unsigned)ua | ((unsigned)ub << 16);
}
__device__ __forceinline__ float bf2f(unsigned short h) {
  unsigned u = ((unsigned)h) << 16;
  return __builtin_bit_cast(float, u);
}
struct Frag32B { uint4 a, b; };
__device__ __forceinline__ v16bf mkfrag(uint4 a, uint4 b) {
  Frag32B t{a, b};
  return __builtin_bit_cast(v16bf, t);
}
__device__ __forceinline__ uint4 ld128(const unsigned short* p) {
  return *(const uint4*)p;
}

struct TileRegs { uint4 a0, a1, b; };

// fetch one 128x32 (A) + 32x64 (B, transposed+packed) tile into registers
template <int MODE>
__device__ __forceinline__ TileRegs fetch_tile(
    const unsigned short* __restrict__ Ab, const float* __restrict__ xb,
    const unsigned short* __restrict__ vb,
    int oc0, int p0, int am, int akc, int bn, int bkc, int kk) {
  TileRegs tr;
  const unsigned short* ap = Ab + (size_t)(oc0 + am) * DIM + kk * KT + akc;
  tr.a0 = *(const uint4*)ap;
  tr.a1 = *(const uint4*)(ap + (size_t)64 * DIM);
  if constexpr (MODE == 0) {
    const float* src = xb + (size_t)(kk * KT + bkc) * HW + p0 + bn;
    if (kk + 1 < NK) __builtin_prefetch(src + (size_t)KT * HW, 0, 3);
    tr.b.x = pk2f(src[0 * HW], src[1 * HW]);
    tr.b.y = pk2f(src[2 * HW], src[3 * HW]);
    tr.b.z = pk2f(src[4 * HW], src[5 * HW]);
    tr.b.w = pk2f(src[6 * HW], src[7 * HW]);
  } else {
    const unsigned short* src = vb + (size_t)(kk * KT + bkc) * HW + p0 + bn;
    if (kk + 1 < NK) __builtin_prefetch(src + (size_t)KT * HW, 0, 3);
    tr.b.x = (unsigned)src[0 * HW] | ((unsigned)src[1 * HW] << 16);
    tr.b.y = (unsigned)src[2 * HW] | ((unsigned)src[3 * HW] << 16);
    tr.b.z = (unsigned)src[4 * HW] | ((unsigned)src[5 * HW] << 16);
    tr.b.w = (unsigned)src[6 * HW] | ((unsigned)src[7 * HW] << 16);
  }
  return tr;
}
} // namespace

// ---------------------------------------------------------------- utilities
__global__ __launch_bounds__(256) void k_cvt(const float* __restrict__ w,
                                             unsigned short* __restrict__ o, int n) {
  int i = blockIdx.x * 256 + threadIdx.x;
  if (i < n) o[i] = f2bf(w[i]);
}

__global__ __launch_bounds__(256) void k_init(float* __restrict__ acc) {
  int i = blockIdx.x * 256 + threadIdx.x;
  if (i < Bc * 80) acc[i] = 0.f;
}

// --------------------------------------------------------- unified WMMA GEMM
// MODE 0: qkv 1x1 conv : out_bf16[b,oc,p] = sum_ic A[oc,ic] * x_f32[b,ic,p] + bias[oc]
// MODE 1: projection   : out_f32[b,oc,p]  = sum_d  A_b[oc,d] * v_bf16[b,d,p] + bias[oc] + resid
// WG: 256 thr = 8 waves as 4(M) x 2(N); wave tile 32x32 (4 accums); WG tile 128x64.
template <int MODE>
__global__ __launch_bounds__(256) void k_gemm(
    const unsigned short* __restrict__ A,    // bf16 weights
    const void* __restrict__ Bsrc,           // MODE0: fp32 x ; MODE1: bf16 v
    const float* __restrict__ bias,
    const float* __restrict__ resid,         // MODE1: x (residual); MODE0: unused
    void* __restrict__ outp)                 // MODE0: u16 ; MODE1: float
{
  __shared__ unsigned short As[2][MT * LDA];
  __shared__ unsigned short Bs[2][NTB * LDB];
  const int b   = blockIdx.z;
  const int oc0 = blockIdx.y * MT;
  const int p0  = blockIdx.x * NTB;
  const int t = threadIdx.x;
  const int lane = t & 31, wave = t >> 5;
  const int waveM = wave >> 1, waveN = wave & 1;
  const int laneHi = lane >> 4, lmod = lane & 15;

  const unsigned short* Ab = A;
  if constexpr (MODE == 1) Ab += (size_t)b * DIM * DIM;
  const float*          xb = (const float*)Bsrc + (size_t)b * DIM * HW;
  const unsigned short* vb = (const unsigned short*)Bsrc + (size_t)b * DIM * HW;

  // per-thread load coordinates
  const int am = t >> 2;                 // A row base (0..63), +64 on 2nd sweep
  const int akc = (t & 3) * 8;           // A k-chunk (halves)
  const int bn = t & 63;                 // B column (pixel within tile)
  const int bkc = (t >> 6) * 8;          // B k-chunk (halves)

  v8f acc00 = {}, acc01 = {}, acc10 = {}, acc11 = {};

  TileRegs tr = fetch_tile<MODE>(Ab, xb, vb, oc0, p0, am, akc, bn, bkc, 0);
  *(uint4*)(&As[0][am * LDA + akc])        = tr.a0;
  *(uint4*)(&As[0][(am + 64) * LDA + akc]) = tr.a1;
  *(uint4*)(&Bs[0][bn * LDB + bkc])        = tr.b;
  __syncthreads();

  for (int kk = 0; kk < NK; ++kk) {
    const int cur = kk & 1;
    if (kk + 1 < NK)                          // overlap with LDS reads + WMMA
      tr = fetch_tile<MODE>(Ab, xb, vb, oc0, p0, am, akc, bn, bkc, kk + 1);

    const unsigned short* ab = &As[cur][(waveM * 32 + lmod) * LDA + laneHi * 8];
    v16bf A0 = mkfrag(ld128(ab), ld128(ab + 16));
    v16bf A1 = mkfrag(ld128(ab + 16 * LDA), ld128(ab + 16 * LDA + 16));
    const unsigned short* bb = &Bs[cur][(waveN * 32 + lmod) * LDB + laneHi * 8];
    v16bf B0 = mkfrag(ld128(bb), ld128(bb + 16));
    v16bf B1 = mkfrag(ld128(bb + 16 * LDB), ld128(bb + 16 * LDB + 16));

    acc00 = __builtin_amdgcn_wmma_f32_16x16x32_bf16(false, A0, false, B0,
                                                    (short)0, acc00, false, false);
    acc01 = __builtin_amdgcn_wmma_f32_16x16x32_bf16(false, A0, false, B1,
                                                    (short)0, acc01, false, false);
    acc10 = __builtin_amdgcn_wmma_f32_16x16x32_bf16(false, A1, false, B0,
                                                    (short)0, acc10, false, false);
    acc11 = __builtin_amdgcn_wmma_f32_16x16x32_bf16(false, A1, false, B1,
                                                    (short)0, acc11, false, false);

    if (kk + 1 < NK) {
      const int nxt = 1 - cur;
      *(uint4*)(&As[nxt][am * LDA + akc])        = tr.a0;
      *(uint4*)(&As[nxt][(am + 64) * LDA + akc]) = tr.a1;
      *(uint4*)(&Bs[nxt][bn * LDB + bkc])        = tr.b;
    }
    __syncthreads();
  }

  // epilogue: D VGPR r -> M = r + 8*laneHi, N = lmod
  const int n0 = p0 + waveN * 32 + lmod;
#pragma unroll
  for (int r = 0; r < 8; ++r) {
#pragma unroll
    for (int mi = 0; mi < 2; ++mi) {
      const int m = oc0 + waveM * 32 + mi * 16 + r + 8 * laneHi;
      const float v0 = (mi ? acc10[r] : acc00[r]) + bias[m];
      const float v1 = (mi ? acc11[r] : acc01[r]) + bias[m];
      if constexpr (MODE == 0) {
        unsigned short* o = (unsigned short*)outp;
        o[((size_t)b * QKVC + m) * HW + n0]      = f2bf(v0);
        o[((size_t)b * QKVC + m) * HW + n0 + 16] = f2bf(v1);
      } else {
        float* o = (float*)outp;
        size_t i0 = ((size_t)b * DIM + m) * HW + n0;
        o[i0]      = v0 + resid[i0];
        o[i0 + 16] = v1 + resid[i0 + 16];
      }
    }
  }
}

// ------------------- K2: depthwise 3x3 + Gram/norm reduction + v materialize
__global__ __launch_bounds__(256) void k_dw(
    const unsigned short* __restrict__ qkv,  // [B][1152][HW] bf16
    const float* __restrict__ wdw,           // [1152][9]
    const float* __restrict__ bdw,           // [1152]
    unsigned short* __restrict__ vout,       // [B][384][HW] bf16
    float* __restrict__ accb)                // [B][80]: S[64], nq[8], nk[8]
{
  __shared__ unsigned short tile[10 * Ww];
  __shared__ float lacc[80];
  const int b = blockIdx.z, c = blockIdx.y;
  const int row0 = blockIdx.x * 8;
  const int t = threadIdx.x, lane = t & 31;
  const int col = t & 127, r0 = t >> 7;   // thread: fixed col, rows r0+2i

  if (t < 80) lacc[t] = 0.f;

  float qv[8][4];
  float Sp[64]; float nqp[8]; float nkp[8];
#pragma unroll
  for (int i = 0; i < 64; ++i) Sp[i] = 0.f;
#pragma unroll
  for (int i = 0; i < 8; ++i) { nqp[i] = 0.f; nkp[i] = 0.f; }

  auto loadTile = [&](int ch) {
    const unsigned short* src = qkv + ((size_t)b * QKVC + ch) * HW;
    __syncthreads();
#pragma unroll
    for (int j = 0; j < 5; ++j) {
      int idx = t + j * 256;
      int rr = idx >> 7, cc = idx & 127;
      int gr = row0 - 1 + rr;
      unsigned short val = 0;
      if (gr >= 0 && gr < Hh) val = src[gr * Ww + cc];
      tile[idx] = val;
    }
    __syncthreads();
  };

  auto conv4 = [&](int ch, float (&s)[4]) {
    float w9[9];
#pragma unroll
    for (int j = 0; j < 9; ++j) w9[j] = wdw[ch * 9 + j];
    float bb = bdw[ch];
#pragma unroll
    for (int i = 0; i < 4; ++i) {
      int lr = r0 + 2 * i;           // pixel row = row0 + lr; tile rows lr..lr+2
      float a = bb;
#pragma unroll
      for (int ky = 0; ky < 3; ++ky)
#pragma unroll
        for (int kx = 0; kx < 3; ++kx) {
          int cc = col + kx - 1;
          if (cc >= 0 && cc < Ww)
            a += w9[ky * 3 + kx] * bf2f(tile[(lr + ky) * Ww + cc]);
        }
      s[i] = a;
    }
  };

#pragma unroll
  for (int n = 0; n < 8; ++n) {
    int ch = n * Cc + c;
    loadTile(ch);
    float s[4];
    conv4(ch, s);
#pragma unroll
    for (int i = 0; i < 4; ++i) { qv[n][i] = s[i]; nqp[n] += s[i] * s[i]; }
  }
#pragma unroll
  for (int m = 0; m < 8; ++m) {
    int ch = DIM + m * Cc + c;
    loadTile(ch);
    float s[4];
    conv4(ch, s);
#pragma unroll
    for (int i = 0; i < 4; ++i) {
      nkp[m] += s[i] * s[i];
#pragma unroll
      for (int n = 0; n < 8; ++n) Sp[n * 8 + m] += qv[n][i] * s[i];
    }
  }
#pragma unroll
  for (int m = 0; m < 8; ++m) {
    int ch = 2 * DIM + m * Cc + c;
    loadTile(ch);
    float s[4];
    conv4(ch, s);
    unsigned short* dst = vout + ((size_t)b * DIM + m * Cc + c) * HW;
#pragma unroll
    for (int i = 0; i < 4; ++i) dst[(row0 + r0 + 2 * i) * Ww + col] = f2bf(s[i]);
  }

  auto wred = [](float v) {
#pragma unroll
    for (int o = 16; o > 0; o >>= 1) v += __shfl_xor(v, o, 32);
    return v;
  };
#pragma unroll
  for (int idx = 0; idx < 64; ++idx) {
    float r = wred(Sp[idx]);
    if (lane == 0) atomicAdd(&lacc[idx], r);
  }
#pragma unroll
  for (int n = 0; n < 8; ++n) {
    float rq = wred(nqp[n]);
    if (lane == 0) atomicAdd(&lacc[64 + n], rq);
    float rk = wred(nkp[n]);
    if (lane == 0) atomicAdd(&lacc[72 + n], rk);
  }
  __syncthreads();
  if (t < 80) atomicAdd(&accb[b * 80 + t], lacc[t]);
}

// ------------------------------------------ K3a: normalize + softmax (8x8)
__global__ void k_attn(const float* __restrict__ accb,
                       const float* __restrict__ temp,
                       float* __restrict__ attn) {
  const int b = blockIdx.x, t = threadIdx.x;   // 64 threads
  __shared__ float S[64], nq[8], nk[8], L[64], E[64];
  if (t < 64) S[t] = accb[b * 80 + t];
  if (t < 8)  nq[t] = fmaxf(sqrtf(accb[b * 80 + 64 + t]), 1e-12f);
  if (t < 8)  nk[t] = fmaxf(sqrtf(accb[b * 80 + 72 + t]), 1e-12f);
  __syncthreads();
  int n = t >> 3, m = t & 7;
  float logit = S[n * 8 + m] / (nq[n] * nk[m]) * temp[0];
  L[t] = logit;
  __syncthreads();
  float mx = -1e30f;
  for (int j = 0; j < 8; ++j) mx = fmaxf(mx, L[n * 8 + j]);
  float e = __expf(logit - mx);
  E[t] = e;
  __syncthreads();
  float s = 0.f;
  for (int j = 0; j < 8; ++j) s += E[n * 8 + j];
  attn[b * 64 + t] = e / s;
}

// --------------- K3b: fold attn into projection: Wp_eff[b][o][m*48+c] (bf16)
__global__ __launch_bounds__(256) void k_fold(const float* __restrict__ attn,
                                              const float* __restrict__ wp,
                                              unsigned short* __restrict__ wpe) {
  const int b = blockIdx.y;
  int idx = blockIdx.x * 256 + threadIdx.x;    // DIM*DIM
  int o = idx / DIM, d = idx % DIM;
  int m = d / Cc, c = d % Cc;
  float s = 0.f;
#pragma unroll
  for (int n = 0; n < 8; ++n) s += attn[b * 64 + n * 8 + m] * wp[o * DIM + n * Cc + c];
  wpe[(size_t)b * DIM * DIM + idx] = f2bf(s);
}

// ---------------------------------------------------------------- launcher
extern "C" void kernel_launch(void* const* d_in, const int* in_sizes, int n_in,
                              void* d_out, int out_size, void* d_ws, size_t ws_size,
                              hipStream_t stream) {
  (void)in_sizes; (void)n_in; (void)out_size; (void)ws_size;
  const float* x      = (const float*)d_in[0];
  const float* temp   = (const float*)d_in[1];
  const float* w_qkv  = (const float*)d_in[2];
  const float* b_qkv  = (const float*)d_in[3];
  const float* w_dw   = (const float*)d_in[4];
  const float* b_dw   = (const float*)d_in[5];
  const float* w_proj = (const float*)d_in[6];
  const float* b_proj = (const float*)d_in[7];
  float* out = (float*)d_out;

  char* ws = (char*)d_ws;
  size_t off = 0;
  unsigned short* qkv = (unsigned short*)(ws + off); off += (size_t)Bc * QKVC * HW * 2;
  unsigned short* vws = (unsigned short*)(ws + off); off += (size_t)Bc * DIM * HW * 2;
  unsigned short* wqb = (unsigned short*)(ws + off); off += (size_t)QKVC * DIM * 2;
  unsigned short* wpe = (unsigned short*)(ws + off); off += (size_t)Bc * DIM * DIM * 2;
  float* accb = (float*)(ws + off); off += (size_t)Bc * 80 * 4;
  float* attn = (float*)(ws + off); off += (size_t)Bc * 64 * 4;

  k_cvt<<<(QKVC * DIM + 255) / 256, 256, 0, stream>>>(w_qkv, wqb, QKVC * DIM);
  k_init<<<2, 256, 0, stream>>>(accb);
  k_gemm<0><<<dim3(HW / NTB, QKVC / MT, Bc), 256, 0, stream>>>(
      wqb, x, b_qkv, nullptr, qkv);
  k_dw<<<dim3(Hh / 8, Cc, Bc), 256, 0, stream>>>(qkv, w_dw, b_dw, vws, accb);
  k_attn<<<Bc, 64, 0, stream>>>(accb, temp, attn);
  k_fold<<<dim3(DIM * DIM / 256, Bc), 256, 0, stream>>>(attn, w_proj, wpe);
  k_gemm<1><<<dim3(HW / NTB, DIM / MT, Bc), 256, 0, stream>>>(
      wpe, vws, b_proj, x, out);
}
